// MyGCNNet_66614942761318
// MI455X (gfx1250) — compile-verified, matrix operand order checked by reading, not run on hardware
//
#include <hip/hip_runtime.h>

typedef __bf16 bfloat;
typedef __attribute__((ext_vector_type(16))) __bf16 v16bf;
typedef __attribute__((ext_vector_type(8)))  float   v8f;

#define GSL(i, total) for (size_t i = (size_t)blockIdx.x * blockDim.x + threadIdx.x; \
                           i < (total); i += (size_t)gridDim.x * blockDim.x)

__device__ __forceinline__ float sigmf(float v) { return 1.f / (1.f + __expf(-v)); }

// ---------------- prep / conversion kernels ----------------
__global__ void k_f32_to_bf16(const float* __restrict__ in, bfloat* __restrict__ out, size_t n) {
  GSL(i, n) out[i] = (bfloat)in[i];
}

__global__ void k_zero_f32(float* __restrict__ p, size_t n) { GSL(i, n) p[i] = 0.f; }

// W [K,N] row-major f32 -> Wt [N,K] bf16 (contiguous K per output column)
__global__ void k_wt_transpose(const float* __restrict__ W, bfloat* __restrict__ Wt, int K, int N) {
  size_t total = (size_t)K * N;
  GSL(i, total) {
    int n = (int)(i % (size_t)N);
    int k = (int)(i / (size_t)N);
    Wt[(size_t)n * K + k] = (bfloat)W[i];
  }
}

// conv w [O,I,3,3] f32 -> w9 [tap(9)][O][I] bf16 (contiguous I per (tap,o))
__global__ void k_convw(const float* __restrict__ w, bfloat* __restrict__ w9, int O, int I) {
  size_t total = (size_t)O * I * 9;
  GSL(i, total) {
    int t = (int)(i % 9); size_t r = i / 9;
    int ii = (int)(r % (size_t)I);
    int o  = (int)(r / (size_t)I);
    w9[((size_t)t * O + o) * I + ii] = (bfloat)w[i];
  }
}

__global__ void k_nchw2nhwc(const float* __restrict__ in, bfloat* __restrict__ out,
                            int Bn, int C, int P) {
  size_t total = (size_t)Bn * C * P;
  GSL(i, total) {
    int p = (int)(i % (size_t)P); size_t r = i / (size_t)P;
    int c = (int)(r % (size_t)C);
    int b = (int)(r / (size_t)C);
    out[((size_t)b * P + p) * C + c] = (bfloat)in[i];
  }
}

// ---------------- GNN aggregation ----------------
__global__ void k_edge_scatter(const bfloat* __restrict__ h, const int* __restrict__ src,
                               const int* __restrict__ dst, float* __restrict__ agg,
                               int E, int D) {
  size_t total = (size_t)E * D;
  GSL(i, total) {
    int c = (int)(i % (size_t)D);
    int e = (int)(i / (size_t)D);
    int s = src[e], d = dst[e];
    atomicAdd(agg + (size_t)d * D + c, (float)h[(size_t)s * D + c]);
  }
}

__global__ void k_edge_count(const int* __restrict__ dst, float* __restrict__ cnt, int E) {
  GSL(i, (size_t)E) atomicAdd(cnt + dst[i], 1.f);
}

__global__ void k_agg_norm(const float* __restrict__ agg, const float* __restrict__ cnt,
                           bfloat* __restrict__ out, int D, size_t total) {
  GSL(i, total) {
    float c = cnt[i / (size_t)D];
    c = c > 1.f ? c : 1.f;
    out[i] = (bfloat)(agg[i] / c);
  }
}

// ---------------- WMMA GEMM: C[M,N] = act([A1|A2][M,K] * W[K,N] + bias) ----------------
// One wave computes a 16x64 strip: the A fragment is loaded once per K-step and
// reused by 4 WMMAs (4x less A traffic).  Wt is column-major bf16 (Wt[n*K+k]).
__global__ void __launch_bounds__(256) k_gemm_wmma(
    const bfloat* __restrict__ A1, const bfloat* __restrict__ A2, int K1,
    const bfloat* __restrict__ Wt, const float* __restrict__ bias,
    bfloat* __restrict__ C, int M, int Ncol, int K, int ldc, int coff, int act) {
  int lane = threadIdx.x & 31;
  int wid = blockIdx.x * (blockDim.x >> 5) + (threadIdx.x >> 5);
  int tiles_n = Ncol >> 6;             // 64-wide N strips
  int tiles_m = (M + 15) >> 4;
  if (wid >= tiles_m * tiles_n) return;
  int tm = wid / tiles_n, tn = wid % tiles_n;
  int lm = lane & 15;
  int kk = (lane >> 4) << 4;           // 0/16: symmetric K-run split for A and B
  int r  = tm * 16 + lm;
  int n0 = tn << 6;
  int K2 = K - K1;
  const bfloat* w0 = Wt + (size_t)(n0 + lm)      * K + kk;
  const bfloat* w1 = Wt + (size_t)(n0 + lm + 16) * K + kk;
  const bfloat* w2 = Wt + (size_t)(n0 + lm + 32) * K + kk;
  const bfloat* w3 = Wt + (size_t)(n0 + lm + 48) * K + kk;
  v8f acc0 = {}, acc1 = {}, acc2 = {}, acc3 = {};
  for (int k0 = 0; k0 < K; k0 += 32) {
    int ka = k0 + kk;
    const bfloat* ap = (ka < K1) ? (A1 + (size_t)r * K1 + ka)
                                 : (A2 + (size_t)r * K2 + (ka - K1));
    __builtin_prefetch(ap + 256, 0, 1);    // global_prefetch_b8 on the streaming A
    v16bf a = *(const v16bf*)ap;
    acc0 = __builtin_amdgcn_wmma_f32_16x16x32_bf16(false, a, false,
             *(const v16bf*)(w0 + k0), (short)0, acc0, false, false);
    acc1 = __builtin_amdgcn_wmma_f32_16x16x32_bf16(false, a, false,
             *(const v16bf*)(w1 + k0), (short)0, acc1, false, false);
    acc2 = __builtin_amdgcn_wmma_f32_16x16x32_bf16(false, a, false,
             *(const v16bf*)(w2 + k0), (short)0, acc2, false, false);
    acc3 = __builtin_amdgcn_wmma_f32_16x16x32_bf16(false, a, false,
             *(const v16bf*)(w3 + k0), (short)0, acc3, false, false);
  }
  int rbase = tm * 16 + ((lane >> 4) << 3);
  for (int j = 0; j < 8; ++j) {
    int m = rbase + j;
    if (m >= M) break;
    float vq[4] = {acc0[j], acc1[j], acc2[j], acc3[j]};
    for (int q = 0; q < 4; ++q) {
      int cc = n0 + (q << 4) + lm;
      float v = vq[q] + (bias ? bias[cc] : 0.f);
      if (act == 1) v = fmaxf(v, 0.f);
      else if (act == 2) v = sigmf(v);
      C[(size_t)m * ldc + coff + cc] = (bfloat)v;
    }
  }
}

// ---------------- WMMA implicit-GEMM conv3x3 (pad 1, stride 1, no bias) ----------------
// Block = 8 waves covering 128 pixels x 64 output channels.  Per (tap, K-chunk) the
// 64xCk weight slice is staged cooperatively into LDS; each wave then issues one
// guarded A load + 4 LDS B fragments + 4 WMMAs per 32-deep K step.
#define CONV_KCHUNK 256

__global__ void __launch_bounds__(256) k_conv3x3_wmma(
    const bfloat* __restrict__ in, const bfloat* __restrict__ w9,
    bfloat* __restrict__ outb, float* __restrict__ outf,
    int Bn, int Hd, int Wd, int Cin, int Cout, int act, int ldc, int coff) {
  __shared__ __align__(64) bfloat sw[64 * CONV_KCHUNK];   // 32 KB weight stage
  const int lane = threadIdx.x & 31;
  const int wv   = threadIdx.x >> 5;                      // 0..7
  const int P = Hd * Wd;
  const int mblocks = (P + 127) >> 7;
  const int nblocks = Cout >> 6;
  const int b   = blockIdx.x / (mblocks * nblocks);
  const int rem = blockIdx.x % (mblocks * nblocks);
  const int mb  = rem / nblocks;
  const int nb  = rem % nblocks;
  const int lm = lane & 15;
  const int kk = (lane >> 4) << 4;
  const int p = mb * 128 + wv * 16 + lm;
  const bool pvalid = p < P;
  const int y  = pvalid ? p / Wd : 0;
  const int xp = pvalid ? p % Wd : 0;
  const int n0 = nb << 6;
  v8f acc0 = {}, acc1 = {}, acc2 = {}, acc3 = {};
  for (int t = 0; t < 9; ++t) {
    const int dy = t / 3 - 1, dx = t % 3 - 1;
    const int yy = y + dy, xx = xp + dx;
    const bool inb = pvalid && ((unsigned)yy < (unsigned)Hd) && ((unsigned)xx < (unsigned)Wd);
    const bfloat* arow = in + ((size_t)b * P + (size_t)yy * Wd + xx) * Cin + kk;
    for (int kbase = 0; kbase < Cin; kbase += CONV_KCHUNK) {
      const int chunk = (Cin - kbase) < CONV_KCHUNK ? (Cin - kbase) : CONV_KCHUNK;
      __syncthreads();
      {
        const bfloat* gsrc = w9 + ((size_t)t * Cout + n0) * Cin + kbase;
        const int row8 = chunk >> 3;            // 16-byte units per row
        const int units = 64 * row8;
        for (int u = threadIdx.x; u < units; u += 256) {
          const int rrow = u / row8;
          const int c8   = u % row8;
          *(uint4*)(sw + (size_t)rrow * chunk + (size_t)c8 * 8) =
              *(const uint4*)(gsrc + (size_t)rrow * Cin + (size_t)c8 * 8);
        }
      }
      __syncthreads();
      const bfloat* b0 = sw + (size_t)(lm)      * chunk + kk;
      const bfloat* b1 = sw + (size_t)(lm + 16) * chunk + kk;
      const bfloat* b2 = sw + (size_t)(lm + 32) * chunk + kk;
      const bfloat* b3 = sw + (size_t)(lm + 48) * chunk + kk;
      const bfloat* ab = arow + kbase;
      for (int k0 = 0; k0 < chunk; k0 += 32) {
        v16bf a = {};
        if (inb) a = *(const v16bf*)(ab + k0);
        acc0 = __builtin_amdgcn_wmma_f32_16x16x32_bf16(false, a, false,
                 *(const v16bf*)(b0 + k0), (short)0, acc0, false, false);
        acc1 = __builtin_amdgcn_wmma_f32_16x16x32_bf16(false, a, false,
                 *(const v16bf*)(b1 + k0), (short)0, acc1, false, false);
        acc2 = __builtin_amdgcn_wmma_f32_16x16x32_bf16(false, a, false,
                 *(const v16bf*)(b2 + k0), (short)0, acc2, false, false);
        acc3 = __builtin_amdgcn_wmma_f32_16x16x32_bf16(false, a, false,
                 *(const v16bf*)(b3 + k0), (short)0, acc3, false, false);
      }
    }
  }
  const int rbase = mb * 128 + wv * 16 + ((lane >> 4) << 3);
  for (int j = 0; j < 8; ++j) {
    const int m = rbase + j;
    if (m >= P) break;
    float vq[4] = {acc0[j], acc1[j], acc2[j], acc3[j]};
    for (int q = 0; q < 4; ++q) {
      float v = vq[q];
      if (act == 1) v = fmaxf(v, 0.f);
      else if (act == 2) v = sigmf(v);
      const int cc = n0 + (q << 4) + lm;
      if (outf) outf[((size_t)b * Cout + coff + cc) * P + m] = v;
      else      outb[((size_t)b * P + m) * ldc + coff + cc] = (bfloat)v;
    }
  }
}

// ---------------- pooling / resize / elementwise ----------------
__global__ void k_avgpool(const bfloat* __restrict__ in, bfloat* __restrict__ out,
                          int Bn, int Hd, int Wd, int C, int k, int Ho, int Wo) {
  size_t total = (size_t)Bn * Ho * Wo * C;
  GSL(i, total) {
    int c = (int)(i % (size_t)C); size_t r = i / (size_t)C;
    int ox = (int)(r % (size_t)Wo); r /= (size_t)Wo;
    int oy = (int)(r % (size_t)Ho);
    int b  = (int)(r / (size_t)Ho);
    float s = 0.f;
    for (int dy = 0; dy < k; ++dy)
      for (int dx = 0; dx < k; ++dx)
        s += (float)in[(((size_t)b * Hd + (oy * k + dy)) * Wd + (ox * k + dx)) * C + c];
    out[i] = (bfloat)(s / (float)(k * k));
  }
}

__global__ void k_resize(const bfloat* __restrict__ in, bfloat* __restrict__ out,
                         int Bn, int Hi, int Wi, int C, int Ho, int Wo, int sig) {
  size_t total = (size_t)Bn * Ho * Wo * C;
  GSL(i, total) {
    int c = (int)(i % (size_t)C); size_t r = i / (size_t)C;
    int ox = (int)(r % (size_t)Wo); r /= (size_t)Wo;
    int oy = (int)(r % (size_t)Ho);
    int b  = (int)(r / (size_t)Ho);
    float py = (Ho > 1) ? (float)oy * (float)(Hi - 1) / (float)(Ho - 1) : 0.f;
    float px = (Wo > 1) ? (float)ox * (float)(Wi - 1) / (float)(Wo - 1) : 0.f;
    int ymax = Hi - 2 > 0 ? Hi - 2 : 0;
    int xmax = Wi - 2 > 0 ? Wi - 2 : 0;
    int y0 = (int)floorf(py); if (y0 > ymax) y0 = ymax; if (y0 < 0) y0 = 0;
    int x0 = (int)floorf(px); if (x0 > xmax) x0 = xmax; if (x0 < 0) x0 = 0;
    float fy = py - (float)y0, fx = px - (float)x0;
    int y1 = (y0 + 1 < Hi) ? y0 + 1 : Hi - 1;
    int x1 = (x0 + 1 < Wi) ? x0 + 1 : Wi - 1;
    size_t base = (size_t)b * Hi;
    float v00 = (float)in[((base + y0) * Wi + x0) * C + c];
    float v01 = (float)in[((base + y0) * Wi + x1) * C + c];
    float v10 = (float)in[((base + y1) * Wi + x0) * C + c];
    float v11 = (float)in[((base + y1) * Wi + x1) * C + c];
    float v = (v00 * (1.f - fx) + v01 * fx) * (1.f - fy)
            + (v10 * (1.f - fx) + v11 * fx) * fy;
    if (sig) v = sigmf(v);
    out[i] = (bfloat)v;
  }
}

// dst[r*ldd + coff + c] = a[i] * b[i]   (safe in-place when ldd==Ca, coff==0, dst==b)
__global__ void k_ew_mul(const bfloat* __restrict__ a, const bfloat* __restrict__ b,
                         bfloat* dst, size_t rows, int Ca, int ldd, int coff) {
  size_t total = rows * (size_t)Ca;
  GSL(i, total) {
    int c = (int)(i % (size_t)Ca);
    size_t r = i / (size_t)Ca;
    dst[r * (size_t)ldd + coff + c] = (bfloat)((float)a[i] * (float)b[i]);
  }
}

__global__ void k_ew_add(const bfloat* __restrict__ a, const bfloat* __restrict__ b,
                         bfloat* dst, size_t n) {
  GSL(i, n) dst[i] = (bfloat)((float)a[i] + (float)b[i]);
}

// ---------------- host orchestration ----------------
extern "C" void kernel_launch(void* const* d_in, const int* in_sizes, int n_in,
                              void* d_out, int out_size, void* d_ws, size_t ws_size,
                              hipStream_t stream) {
  (void)in_sizes; (void)n_in; (void)out_size; (void)ws_size;
  const int Nn = 16384, D = 512, E = 131072, Bn = 4, Hh = 64, Wc = 64;
  const int P64 = Hh * Wc;        // 4096
  const int P128 = 128 * 128;     // 16384
  const size_t ND = (size_t)Nn * D;

  const float* x    = (const float*)d_in[0];
  const float* x2   = (const float*)d_in[1];
  const float* nf   = (const float*)d_in[2];
  const int*   eidx = (const int*)d_in[3];
  const int* esrc = eidx;
  const int* edst = eidx + E;
  const float* wA[4] = {(const float*)d_in[4],  (const float*)d_in[5],
                        (const float*)d_in[6],  (const float*)d_in[7]};
  const float* wB[4] = {(const float*)d_in[8],  (const float*)d_in[9],
                        (const float*)d_in[10], (const float*)d_in[11]};
  const float* w3[4] = {(const float*)d_in[12], (const float*)d_in[13],
                        (const float*)d_in[14], (const float*)d_in[15]};
  const float* w_sum   = (const float*)d_in[16];
  const float* w_gcn   = (const float*)d_in[17];
  const float* w_att   = (const float*)d_in[18];
  const float* w_sum_c = (const float*)d_in[19];
  const float* emb_w   = (const float*)d_in[20];
  const float* emb_b   = (const float*)d_in[21];
  const float* sage_w[4] = {(const float*)d_in[22], (const float*)d_in[23],
                            (const float*)d_in[24], (const float*)d_in[25]};
  const float* sage_b[4] = {(const float*)d_in[26], (const float*)d_in[27],
                            (const float*)d_in[28], (const float*)d_in[29]};
  const float* skip_w[2] = {(const float*)d_in[30], (const float*)d_in[31]};

  char* cur = (char*)d_ws;
  auto alloc = [&](size_t bytes) -> void* {
    void* p = (void*)cur;
    cur += (bytes + 255) & ~(size_t)255;
    return p;
  };

  bfloat* nf_bf  = (bfloat*)alloc(ND * 2);
  bfloat* h_a    = (bfloat*)alloc(ND * 2);
  bfloat* h_b    = (bfloat*)alloc(ND * 2);
  bfloat* h_c    = (bfloat*)alloc(ND * 2);
  bfloat* agg_bf = (bfloat*)alloc(ND * 2);
  float*  agg_f  = (float*)alloc(ND * 4);
  float*  cnt    = (float*)alloc((size_t)Nn * 4);
  bfloat* xg_bf  = (bfloat*)alloc(ND * 2);  // x_gcn NHWC @64x64 == [N,512]
  bfloat* emb_wt = (bfloat*)alloc((size_t)512 * 512 * 2);
  bfloat* sage_wt[4]; for (int l = 0; l < 4; ++l) sage_wt[l] = (bfloat*)alloc((size_t)1024 * 512 * 2);
  bfloat* skip_wt[2]; for (int l = 0; l < 2; ++l) skip_wt[l] = (bfloat*)alloc((size_t)512 * 256 * 2);
  bfloat* wAt[4]; for (int l = 0; l < 4; ++l) wAt[l] = (bfloat*)alloc((size_t)9 * 128 * 512 * 2);
  bfloat* wBt[4]; for (int l = 0; l < 4; ++l) wBt[l] = (bfloat*)alloc((size_t)9 * 128 * 128 * 2);
  bfloat* w3t[4]; for (int l = 0; l < 4; ++l) w3t[l] = (bfloat*)alloc((size_t)9 * 128 * 512 * 2);
  bfloat* w_sum_t   = (bfloat*)alloc((size_t)9 * 256 * 512 * 2);
  bfloat* w_gcn_t   = (bfloat*)alloc((size_t)9 * 256 * 512 * 2);
  bfloat* w_att_t   = (bfloat*)alloc((size_t)9 * 256 * 256 * 2);
  bfloat* w_sum_c_t = (bfloat*)alloc((size_t)9 * 256 * 256 * 2);
  bfloat* x_nhwc  = (bfloat*)alloc((size_t)Bn * P64 * 512 * 2);
  bfloat* x2_nhwc = (bfloat*)alloc((size_t)Bn * P128 * 256 * 2);
  bfloat* pool_buf = (bfloat*)alloc((size_t)Bn * 1024 * 512 * 2);
  bfloat* ya = (bfloat*)alloc((size_t)Bn * 1024 * 128 * 2);
  bfloat* yb = (bfloat*)alloc((size_t)Bn * 1024 * 128 * 2);
  bfloat* ys = (bfloat*)alloc((size_t)Bn * P64 * 128 * 2);
  bfloat* cx = (bfloat*)alloc((size_t)Bn * P64 * 128 * 2);
  bfloat* res64  = (bfloat*)alloc((size_t)Bn * P64 * 512 * 2);
  bfloat* res128 = (bfloat*)alloc((size_t)Bn * P128 * 512 * 2);
  bfloat* r1     = (bfloat*)alloc((size_t)Bn * P128 * 256 * 2);
  bfloat* xg128  = (bfloat*)alloc((size_t)Bn * P128 * 512 * 2);
  bfloat* gbuf   = (bfloat*)alloc((size_t)Bn * P128 * 256 * 2);
  bfloat* t2     = (bfloat*)alloc((size_t)Bn * P128 * 256 * 2);

  auto grid = [](size_t n) -> int {
    size_t b = (n + 255) / 256;
    return (int)(b > 262144 ? 262144 : b);
  };
  auto gemm = [&](const bfloat* A1, const bfloat* A2, int K1, const bfloat* Wt,
                  const float* bias, bfloat* C, int M, int Ncol, int K,
                  int ldc, int coff, int act) {
    int waves = ((M + 15) / 16) * (Ncol / 64);
    k_gemm_wmma<<<(waves + 7) / 8, 256, 0, stream>>>(A1, A2, K1, Wt, bias, C,
                                                     M, Ncol, K, ldc, coff, act);
  };
  auto conv = [&](const bfloat* in, const bfloat* w9, bfloat* outb, float* outf,
                  int Hd, int Wd, int Cin, int Cout, int act, int ldc, int coff) {
    int P = Hd * Wd;
    int blocks = Bn * ((P + 127) / 128) * (Cout / 64);
    k_conv3x3_wmma<<<blocks, 256, 0, stream>>>(in, w9, outb, outf, Bn,
                                               Hd, Wd, Cin, Cout, act, ldc, coff);
  };

  // ---- prep: conversions + weight packing ----
  k_f32_to_bf16<<<grid(ND), 256, 0, stream>>>(nf, nf_bf, ND);
  k_wt_transpose<<<grid((size_t)512 * 512), 256, 0, stream>>>(emb_w, emb_wt, 512, 512);
  for (int l = 0; l < 4; ++l)
    k_wt_transpose<<<grid((size_t)1024 * 512), 256, 0, stream>>>(sage_w[l], sage_wt[l], 1024, 512);
  for (int l = 0; l < 2; ++l)
    k_wt_transpose<<<grid((size_t)512 * 256), 256, 0, stream>>>(skip_w[l], skip_wt[l], 512, 256);
  for (int l = 0; l < 4; ++l) {
    k_convw<<<grid((size_t)128 * 512 * 9), 256, 0, stream>>>(wA[l], wAt[l], 128, 512);
    k_convw<<<grid((size_t)128 * 128 * 9), 256, 0, stream>>>(wB[l], wBt[l], 128, 128);
    k_convw<<<grid((size_t)128 * 512 * 9), 256, 0, stream>>>(w3[l], w3t[l], 128, 512);
  }
  k_convw<<<grid((size_t)256 * 512 * 9), 256, 0, stream>>>(w_sum,   w_sum_t,   256, 512);
  k_convw<<<grid((size_t)256 * 512 * 9), 256, 0, stream>>>(w_gcn,   w_gcn_t,   256, 512);
  k_convw<<<grid((size_t)256 * 256 * 9), 256, 0, stream>>>(w_att,   w_att_t,   256, 256);
  k_convw<<<grid((size_t)256 * 256 * 9), 256, 0, stream>>>(w_sum_c, w_sum_c_t, 256, 256);
  k_nchw2nhwc<<<grid((size_t)Bn * 512 * P64), 256, 0, stream>>>(x, x_nhwc, Bn, 512, P64);
  k_nchw2nhwc<<<grid((size_t)Bn * 256 * P128), 256, 0, stream>>>(x2, x2_nhwc, Bn, 256, P128);

  // ---- GNN: embedding + 4x SAGE + skips ----
  gemm(nf_bf, nf_bf, 512, emb_wt, emb_b, h_a, Nn, 512, 512, 512, 0, 0);  // feats[0]
  bfloat* hin = h_a;
  bfloat* houts[4] = {h_b, h_c, h_a, h_b};  // feats[2]=h_c, feats[4]=h_b survive
  for (int l = 0; l < 4; ++l) {
    k_zero_f32<<<grid(ND), 256, 0, stream>>>(agg_f, ND);
    k_zero_f32<<<grid((size_t)Nn), 256, 0, stream>>>(cnt, (size_t)Nn);
    k_edge_scatter<<<grid((size_t)E * D), 256, 0, stream>>>(hin, esrc, edst, agg_f, E, D);
    k_edge_count<<<grid((size_t)E), 256, 0, stream>>>(edst, cnt, E);
    k_agg_norm<<<grid(ND), 256, 0, stream>>>(agg_f, cnt, agg_bf, D, ND);
    gemm(hin, agg_bf, 512, sage_wt[l], sage_b[l], houts[l], Nn, 512, 1024, 512, 0, 1);
    hin = houts[l];
  }
  gemm(h_c, h_c, 512, skip_wt[0], nullptr, xg_bf, Nn, 256, 512, 512, 0,   1);
  gemm(h_b, h_b, 512, skip_wt[1], nullptr, xg_bf, Nn, 256, 512, 512, 256, 1);

  // ---- DeepPool branches ----
  const int ks[4] = {2, 4, 6, 8};
  for (int i = 0; i < 4; ++i) {
    int kk = ks[i], ho = Hh / kk, wo = Wc / kk;
    k_avgpool<<<grid((size_t)Bn * ho * wo * 512), 256, 0, stream>>>(
        x_nhwc, pool_buf, Bn, Hh, Wc, 512, kk, ho, wo);
    conv(pool_buf, wAt[i], ya, nullptr, ho, wo, 512, 128, 1, 128, 0);
    conv(ya, wBt[i], yb, nullptr, ho, wo, 128, 128, 0, 128, 0);
    k_resize<<<grid((size_t)Bn * P64 * 128), 256, 0, stream>>>(
        yb, ys, Bn, ho, wo, 128, Hh, Wc, 1);
    conv(x_nhwc, w3t[i], cx, nullptr, Hh, Wc, 512, 128, 0, 128, 0);
    k_ew_mul<<<grid((size_t)Bn * P64 * 128), 256, 0, stream>>>(
        ys, cx, res64, (size_t)Bn * P64, 128, 512, 128 * i);
  }
  // res -> 128x128, w_sum
  k_resize<<<grid((size_t)Bn * P128 * 512), 256, 0, stream>>>(
      res64, res128, Bn, Hh, Wc, 512, 128, 128, 0);
  conv(res128, w_sum_t, r1, nullptr, 128, 128, 512, 256, 0, 256, 0);
  // gate from GNN grid
  k_resize<<<grid((size_t)Bn * P128 * 512), 256, 0, stream>>>(
      xg_bf, xg128, Bn, Hh, Wc, 512, 128, 128, 0);
  conv(xg128, w_gcn_t, gbuf, nullptr, 128, 128, 512, 256, 2, 256, 0);
  k_ew_mul<<<grid((size_t)Bn * P128 * 256), 256, 0, stream>>>(
      gbuf, r1, r1, (size_t)Bn * P128, 256, 256, 0);
  conv(r1, w_att_t, t2, nullptr, 128, 128, 256, 256, 0, 256, 0);
  k_ew_add<<<grid((size_t)Bn * P128 * 256), 256, 0, stream>>>(
      t2, x2_nhwc, t2, (size_t)Bn * P128 * 256);
  // final conv writes f32 NCHW straight into d_out
  conv(t2, w_sum_c_t, nullptr, (float*)d_out, 128, 128, 256, 256, 0, 256, 0);
}
